// WeightOnlyInt4Linear_38431367364720
// MI455X (gfx1250) — compile-verified, matrix operand order checked by reading, not run on hardware
//
#include <hip/hip_runtime.h>

typedef __attribute__((ext_vector_type(16))) __bf16 v16bf;
typedef __attribute__((ext_vector_type(8)))  float  v8f;
typedef __attribute__((ext_vector_type(4)))  float  f32x4;
typedef __attribute__((ext_vector_type(4)))  int    i32x4;

constexpr int O_DIM  = 11008;   // output features
constexpr int K_DIM  = 4096;    // reduction dim
constexpr int N_ROWS = 8192;    // B*S flattened rows
constexpr int BM = 128, BN = 128, BK = 64;
constexpr int LDSLD = 80;       // padded LDS row stride (elements): 160B, keeps 32B frags aligned
constexpr int KT = K_DIM / BK;  // 64 k-tiles
constexpr int ATILE = BM * LDSLD;            // elements per A buffer
constexpr int BTILE = BN * LDSLD;            // elements per B buffer
constexpr size_t LDS_BYTES = (size_t)(2 * ATILE + 2 * BTILE) * sizeof(__bf16); // 81920

// Pack bf16(lo) | bf16(hi)<<16 with one v_perm_b32 after a round-bias add.
static __device__ __forceinline__ unsigned pk2(float lo, float hi) {
  unsigned ul = __builtin_bit_cast(unsigned, lo) + 0x8000u;  // round-to-nearest (ties away)
  unsigned uh = __builtin_bit_cast(unsigned, hi) + 0x8000u;
  return __builtin_amdgcn_perm(uh, ul, 0x07060302u);         // {uh[31:16], ul[31:16]}
}
static __device__ __forceinline__ float bf2f(unsigned short h) {
  return __builtin_bit_cast(float, ((unsigned)h) << 16);
}

__global__ __launch_bounds__(256)
void wo_int4_gemm_bf16_wmma(const float* __restrict__ x,
                            const int* __restrict__ wq,
                            const unsigned short* __restrict__ sz,  // [G][O][2] bf16 bits
                            float* __restrict__ out)
{
  extern __shared__ __align__(32) __bf16 smem[];
  __bf16* const Abase = smem;                 // [2][BM][LDSLD]
  __bf16* const Bbase = smem + 2 * ATILE;     // [2][BN][LDSLD]

  const int tid  = threadIdx.x;
  const int colb = blockIdx.x * BN;   // output-feature base
  const int rowb = blockIdx.y * BM;   // row base

  // cooperative load coords: each thread owns 32 contiguous K of one row
  const int lr = tid >> 1;            // 0..127
  const int lk = (tid & 1) * 32;      // 0 or 32
  const float* xsrc = x  + (size_t)(rowb + lr) * K_DIM + lk;
  const int*   wsrc = wq + (size_t)(colb + lr) * K_DIM + lk;
  const unsigned short* szsrc = sz + (size_t)(colb + lr) * 2;

  // wave coords: 4 waves in M x 2 waves in N; each wave does 32x64
  const int wave = tid >> 5;
  const int lane = tid & 31;
  const int wm   = (wave & 3) * 32;
  const int wn   = (wave >> 2) * 64;
  const int half = lane >> 4;         // K-half selector (fragment layout)
  const int hidx = lane & 15;         // M (for A) / N (for B,C,D) index

  v8f acc[2][4] = {};

  f32x4 xr[8];
  i32x4 wr[8];
  float s = 0.f, z2 = 0.f;            // z2 = zero - 8*scale (folds the int4 offset)

  auto fetch = [&](int kt) {
    const f32x4* xp = (const f32x4*)(xsrc + (size_t)kt * BK);
#pragma unroll
    for (int i = 0; i < 8; ++i) xr[i] = __builtin_nontemporal_load(xp + i);  // x streams: NT
    const i32x4* wp = (const i32x4*)(wsrc + (size_t)kt * BK);
#pragma unroll
    for (int i = 0; i < 8; ++i) wr[i] = wp[i];                               // weights: RT, L2-resident
    if ((kt & 1) == 0) {               // group-128 boundary: only every other 64-wide tile
      const int g = kt >> 1;
      s  = bf2f(szsrc[(size_t)g * O_DIM * 2 + 0]);
      z2 = bf2f(szsrc[(size_t)g * O_DIM * 2 + 1]) - 8.0f * s;
    }
  };

  auto stage = [&](int buf) {         // regs -> LDS: convert x, dequantize weights
    unsigned* ap = (unsigned*)(Abase + buf * ATILE + lr * LDSLD + lk);
#pragma unroll
    for (int i = 0; i < 8; ++i) {
      ap[2 * i + 0] = pk2(xr[i][0], xr[i][1]);
      ap[2 * i + 1] = pk2(xr[i][2], xr[i][3]);
    }
    unsigned* bp = (unsigned*)(Bbase + buf * BTILE + lr * LDSLD + lk);
#pragma unroll
    for (int i = 0; i < 8; ++i) {
      const i32x4 q = wr[i];
      const float w0 = (float)q[0] * s + z2;   // v_cvt_f32_i32 + v_fmac
      const float w1 = (float)q[1] * s + z2;
      const float w2 = (float)q[2] * s + z2;
      const float w3 = (float)q[3] * s + z2;
      bp[2 * i + 0] = pk2(w0, w1);
      bp[2 * i + 1] = pk2(w2, w3);
    }
  };

  fetch(0);
  stage(0);
  __syncthreads();

  for (int kt = 0; kt < KT; ++kt) {
    const int cur  = kt & 1;
    const bool more = (kt + 1) < KT;

    if (more) fetch(kt + 1);          // global loads overlap this tile's WMMAs

    // compute: 16 WMMAs per wave per k-tile
    const __bf16* Ab = Abase + cur * ATILE;
    const __bf16* Bb = Bbase + cur * BTILE;
#pragma unroll
    for (int kk = 0; kk < BK; kk += 32) {
      const int ks = kk + half * 16;
      v16bf a0 = *(const v16bf*)(Ab + (wm + 0  + hidx) * LDSLD + ks);
      v16bf a1 = *(const v16bf*)(Ab + (wm + 16 + hidx) * LDSLD + ks);
      v16bf bfr[4];
#pragma unroll
      for (int nt = 0; nt < 4; ++nt)
        bfr[nt] = *(const v16bf*)(Bb + (wn + nt * 16 + hidx) * LDSLD + ks);
#pragma unroll
      for (int nt = 0; nt < 4; ++nt) {
        acc[0][nt] = __builtin_amdgcn_wmma_f32_16x16x32_bf16(
            false, a0, false, bfr[nt], (short)0, acc[0][nt], false, false);
        acc[1][nt] = __builtin_amdgcn_wmma_f32_16x16x32_bf16(
            false, a1, false, bfr[nt], (short)0, acc[1][nt], false, false);
      }
    }

    if (more) stage(cur ^ 1);         // staging VALU/ds_store co-executes with WMMA pipe
    __syncthreads();                  // single barrier per k-tile
  }

  // epilogue: D layout = VGPR r, lane l: M = r + 8*(l/16), N = l%16
#pragma unroll
  for (int mt = 0; mt < 2; ++mt) {
#pragma unroll
    for (int nt = 0; nt < 4; ++nt) {
      const int o = colb + wn + nt * 16 + hidx;
#pragma unroll
      for (int r = 0; r < 8; ++r) {
        const int m = rowb + wm + mt * 16 + r + 8 * half;
        __builtin_nontemporal_store(acc[mt][nt][r], out + (size_t)m * O_DIM + o);
      }
    }
  }
}

extern "C" void kernel_launch(void* const* d_in, const int* in_sizes, int n_in,
                              void* d_out, int out_size, void* d_ws, size_t ws_size,
                              hipStream_t stream) {
  (void)in_sizes; (void)n_in; (void)out_size; (void)d_ws; (void)ws_size;
  const float*          xx = (const float*)d_in[0];
  const int*            wq = (const int*)d_in[1];
  const unsigned short* sz = (const unsigned short*)d_in[2];  // bf16 bits
  float*                out = (float*)d_out;

  dim3 grid(O_DIM / BN, N_ROWS / BM);   // (86, 64)
  wo_int4_gemm_bf16_wmma<<<grid, 256, LDS_BYTES, stream>>>(xx, wq, sz, out);
}
